// BANLayer_62294205661461
// MI455X (gfx1250) — compile-verified, matrix operand order checked by reading, not run on hardware
//
#include <hip/hip_runtime.h>
#include <hip/hip_bf16.h>

typedef __attribute__((ext_vector_type(16))) _Float16 v16h;
typedef __attribute__((ext_vector_type(8)))  _Float16 h8;
typedef __attribute__((ext_vector_type(8)))  float    v8f;
typedef __attribute__((ext_vector_type(4)))  float    f4;

#define BB 64
#define NN 512
#define MM 1024
#define DD 128

// ---- fragment fill helpers (16-bit A/B WMMA layout, ISA 7.12.2) ----
// For lane: lo = lane&15 selects row (A) / col (B); hi = lane>>4 selects the
// 8-wide K sub-block.  Element e in v16h: K = k0 + hi*8 + (e&7) + (e>>3)*16.

__device__ __forceinline__ void ld8_f32_h(const float* __restrict__ p, v16h& a, int base) {
  f4 x0 = *(const f4*)p;
  f4 x1 = *(const f4*)(p + 4);
#pragma unroll
  for (int i = 0; i < 4; ++i) {
    a[base + i]     = (_Float16)x0[i];
    a[base + 4 + i] = (_Float16)x1[i];
  }
}

__device__ __forceinline__ void ld8_h(const _Float16* __restrict__ p, v16h& a, int base) {
  h8 x = *(const h8*)p;
#pragma unroll
  for (int i = 0; i < 8; ++i) a[base + i] = x[i];
}

// ---- Kernel 1: Y = relu(X @ W^T + b) stored as f16 (optionally also Y^T).
//      X:[rows,128], W:[128,128].  YT (if non-null) is [b][d][m] with batch-
//      local m (rows = BB*MM); used so the context GEMM can read Vp
//      column-major with contiguous 16-byte loads.
__global__ __launch_bounds__(256) void ban_proj_relu_f16(
    const float* __restrict__ X, const float* __restrict__ W,
    const float* __restrict__ bias, _Float16* __restrict__ Y,
    _Float16* __restrict__ YT, int rows)
{
  const int lane = threadIdx.x & 31;
  const int wid  = threadIdx.x >> 5;
  const int gw   = blockIdx.x * 8 + wid;          // global wave id
  const int rowTile = gw >> 3;                    // 8 col-tiles of 16 across D=128
  const int colTile = gw & 7;
  if (rowTile * 16 >= rows) return;               // wave-uniform guard

  const int lo = lane & 15, hi = lane >> 4;
  const float* xrow = X + (size_t)(rowTile * 16 + lo) * DD;
  const float* wcol = W + (size_t)(colTile * 16 + lo) * DD;   // B[k][n] = W[n][k]

  v8f acc = {};
#pragma unroll
  for (int kk = 0; kk < 4; ++kk) {
    const int k0 = kk * 32 + hi * 8;
    v16h a, b;
    ld8_f32_h(xrow + k0,      a, 0);
    ld8_f32_h(xrow + k0 + 16, a, 8);
    ld8_f32_h(wcol + k0,      b, 0);
    ld8_f32_h(wcol + k0 + 16, b, 8);
    acc = __builtin_amdgcn_wmma_f32_16x16x32_f16(false, a, false, b,
                                                 (short)0, acc, false, false);
  }

  const int col = colTile * 16 + lo;
  const float bv = bias[col];
  _Float16 yh[8];
#pragma unroll
  for (int v = 0; v < 8; ++v) {
    const int r = rowTile * 16 + v + hi * 8;      // D-matrix: M = v + hi*8, N = lo
    float y = acc[v] + bv;
    y = y > 0.f ? y : 0.f;
    yh[v] = (_Float16)y;
    Y[(size_t)r * DD + col] = yh[v];
  }
  if (YT != nullptr) {
    // rows == BB*MM here; tiles never straddle batches (MM % 16 == 0)
    const int bb     = rowTile >> 6;              // / (MM/16)
    const int m_base = (rowTile & 63) * 16 + hi * 8;
    _Float16* tcol = YT + ((size_t)bb * DD + col) * MM + m_base;
#pragma unroll
    for (int v = 0; v < 8; ++v) tcol[v] = yh[v];
  }
}

// ---- Kernel 2: attn + softmax + context + joint-mean, one block per (b, n-tile) ----
__global__ __launch_bounds__(256) void ban_attn_ctx(
    const _Float16* __restrict__ Uc, const _Float16* __restrict__ Vp,
    const _Float16* __restrict__ VpT, const float* __restrict__ q,
    float* __restrict__ out)
{
  __shared__ float sS[16 * MM];       // 64 KB score/prob matrix for this n-tile
  __shared__ float sRed[16 * 16];
  __shared__ float sRowMax[16];
  __shared__ float sRowInv[16];

  const int b   = blockIdx.x >> 5;    // N/16 = 32 n-tiles per batch
  const int nt  = blockIdx.x & 31;
  const int tid = threadIdx.x;
  const int lane = tid & 31, wid = tid >> 5;
  const int lo = lane & 15, hi = lane >> 4;

  // ---- Uc A-fragments for rows nt*16..nt*16+15, K = D = 128 (shared by all m-tiles)
  const _Float16* ucRow = Uc + ((size_t)b * NN + nt * 16 + lo) * DD;
  v16h aU[4];
#pragma unroll
  for (int kk = 0; kk < 4; ++kk) {
    const int k0 = kk * 32 + hi * 8;
    ld8_h(ucRow + k0,      aU[kk], 0);
    ld8_h(ucRow + k0 + 16, aU[kk], 8);
  }

  // ---- Phase A: attn = Uc @ Vp^T ; wave handles 8 m-tiles -> LDS
  const _Float16* VpB = Vp + (size_t)b * MM * DD;
#pragma unroll 1
  for (int t = 0; t < 8; ++t) {
    const int mt = wid * 8 + t;
    const _Float16* vrow = VpB + (size_t)(mt * 16 + lo) * DD;   // B[k=d][n=m]
    v8f acc = {};
#pragma unroll
    for (int kk = 0; kk < 4; ++kk) {
      const int k0 = kk * 32 + hi * 8;
      v16h bfr;
      ld8_h(vrow + k0,      bfr, 0);
      ld8_h(vrow + k0 + 16, bfr, 8);
      acc = __builtin_amdgcn_wmma_f32_16x16x32_f16(false, aU[kk], false, bfr,
                                                   (short)0, acc, false, false);
    }
#pragma unroll
    for (int v = 0; v < 8; ++v)
      sS[(v + hi * 8) * MM + mt * 16 + lo] = acc[v];
  }
  __syncthreads();

  // ---- Phase B: softmax over M, 16 threads per row, 64 elems each
  const int srow = tid & 15;
  const int schunk = tid >> 4;
  float* rowp = sS + srow * MM + schunk * 64;
  float mx = -3.4e38f;
#pragma unroll 4
  for (int i = 0; i < 64; ++i) mx = fmaxf(mx, rowp[i]);
  sRed[srow * 16 + schunk] = mx;
  __syncthreads();
  if (schunk == 0) {
    float m = sRed[srow * 16];
#pragma unroll
    for (int i = 1; i < 16; ++i) m = fmaxf(m, sRed[srow * 16 + i]);
    sRowMax[srow] = m;
  }
  __syncthreads();
  const float rm = sRowMax[srow];
  float s = 0.f;
#pragma unroll 4
  for (int i = 0; i < 64; ++i) { float e = __expf(rowp[i] - rm); rowp[i] = e; s += e; }
  sRed[srow * 16 + schunk] = s;
  __syncthreads();
  if (schunk == 0) {
    float tsum = 0.f;
#pragma unroll
    for (int i = 0; i < 16; ++i) tsum += sRed[srow * 16 + i];
    sRowInv[srow] = 1.0f / tsum;
  }
  __syncthreads();
  const float inv = sRowInv[srow];
#pragma unroll 4
  for (int i = 0; i < 64; ++i) rowp[i] *= inv;
  __syncthreads();

  // ---- Phase C: context = P(16x1024) @ Vp(1024x128); wave owns d-tile `wid`.
  //      B-frags read the transposed copy VpT[b][d][m]: contiguous 16B runs.
  const int d = wid * 16 + lo;
  const _Float16* vcolT = VpT + ((size_t)b * DD + d) * MM;
  v8f acc2 = {};
#pragma unroll 1
  for (int km = 0; km < 32; ++km) {
    const int k0 = km * 32 + hi * 8;
    v16h aP, bV;
    const float* pr = sS + lo * MM + k0;          // A[row=lo][k=m]
#pragma unroll
    for (int i = 0; i < 8; ++i) {
      aP[i]     = (_Float16)pr[i];
      aP[8 + i] = (_Float16)pr[16 + i];
    }
    ld8_h(vcolT + k0,      bV, 0);                // B[k=m][n=d]
    ld8_h(vcolT + k0 + 16, bV, 8);
    acc2 = __builtin_amdgcn_wmma_f32_16x16x32_f16(false, aP, false, bV,
                                                  (short)0, acc2, false, false);
  }

  // ---- Epilogue: joint = (Uc + context)*q, accumulate mean over N
  const float qv = q[d];
  const _Float16* ucCol = Uc + ((size_t)b * NN + nt * 16) * DD + d;
  float partial = 0.f;
#pragma unroll
  for (int v = 0; v < 8; ++v) {
    const int nl = v + hi * 8;
    const float u = (float)ucCol[(size_t)nl * DD];
    partial += (acc2[v] + u) * qv;
  }
  atomicAdd(out + b * DD + d, partial * (1.0f / (float)NN));
}

__global__ void ban_zero(float* __restrict__ o, int n) {
  int i = blockIdx.x * blockDim.x + threadIdx.x;
  if (i < n) o[i] = 0.f;
}

extern "C" void kernel_launch(void* const* d_in, const int* in_sizes, int n_in,
                              void* d_out, int out_size, void* d_ws, size_t ws_size,
                              hipStream_t stream) {
  const float* h_c = (const float*)d_in[0];   // (B,N,D)
  const float* h_p = (const float*)d_in[1];   // (B,M,D)
  const float* U_w = (const float*)d_in[2];   // (D,D)
  const float* U_b = (const float*)d_in[3];   // (D,)
  const float* V_w = (const float*)d_in[4];   // (D,D)
  const float* V_b = (const float*)d_in[5];   // (D,)
  const float* q   = (const float*)d_in[6];   // (D,)
  float* out = (float*)d_out;                 // (B,D)

  _Float16* Uc  = (_Float16*)d_ws;                        // B*N*D halves  =  8 MB
  _Float16* Vp  = Uc  + (size_t)BB * NN * DD;             // B*M*D halves  = 16 MB
  _Float16* VpT = Vp  + (size_t)BB * MM * DD;             // B*D*M halves  = 16 MB

  // Projections (one wave per 16x16 tile; 8 waves/block)
  {
    const int rows_c = BB * NN;                 // 32768
    const int waves  = (rows_c / 16) * (DD / 16);
    ban_proj_relu_f16<<<waves / 8, 256, 0, stream>>>(h_c, U_w, U_b, Uc, nullptr, rows_c);
  }
  {
    const int rows_p = BB * MM;                 // 65536
    const int waves  = (rows_p / 16) * (DD / 16);
    ban_proj_relu_f16<<<waves / 8, 256, 0, stream>>>(h_p, V_w, V_b, Vp, VpT, rows_p);
  }

  // Zero output (harness poisons d_out; we accumulate with atomics)
  ban_zero<<<(BB * DD + 255) / 256, 256, 0, stream>>>(out, BB * DD);

  // Fused attention + softmax + context + joint mean
  ban_attn_ctx<<<BB * (NN / 16), 256, 0, stream>>>(Uc, Vp, VpT, q, out);
}
